// MultiHeaded_Sent_Attention_44727789420584
// MI455X (gfx1250) — matrix operand order repaired and checked by reading, not compiled, not run on hardware
//
#include <hip/hip_runtime.h>
#include <hip/hip_bf16.h>
#include <math.h>

// ---------------------------------------------------------------------------
// MI455X (gfx1250, wave32) fused multi-head "sentence" attention.
// B=8, L=64, D=768, H=12, d=64.  BI = B*L = 512 query rows.
// ---------------------------------------------------------------------------

#define LL 64
#define DD 768
#define HH 12
#define BI 512

typedef __attribute__((ext_vector_type(16))) __bf16 v16bf;
typedef __attribute__((ext_vector_type(8)))  float  v8f;

union BF16Frag {
    v16bf v;
    unsigned int u[8];
    unsigned short s[16];
};

// fp32 -> bf16 using the native convert (lowers to v_cvt_*bf16* on gfx1250)
__device__ __forceinline__ unsigned short f2bf(float f) {
    union { __bf16 b; unsigned short s; } t;
    t.b = (__bf16)f;
    return t.s;
}
__device__ __forceinline__ unsigned int pack2bf(float a, float b) {
    union { __bf16 h[2]; unsigned int u; } t;
    t.h[0] = (__bf16)a;     // K even -> [15:0]
    t.h[1] = (__bf16)b;     // K odd  -> [31:16]
    return t.u;
}

// ---------------------------------------------------------------------------
// Kernel 1/4: out[M=512, N=768] = (A[512,768] @ W[768,768] + bias) * scale
// One wave per 16x16 C tile.  grid = (32, 48), block = 32.
// ---------------------------------------------------------------------------
__global__ __launch_bounds__(32) void gemm512_bias(
    const float* __restrict__ A, const float* __restrict__ W,
    const float* __restrict__ bias, float* __restrict__ out, float scale)
{
    const int m0 = blockIdx.x * 16;
    const int n0 = blockIdx.y * 16;
    const int lane = threadIdx.x & 31;
    const int lhi = (lane >> 4) & 1;   // 0: lanes 0-15, 1: lanes 16-31
    const int l16 = lane & 15;

    v8f acc = {};
    for (int k0 = 0; k0 < DD; k0 += 32) {
        BF16Frag Af, Bf;
        // A fragment: 16x32 bf16, M = l16, K layout per ISA table
        const int arow = m0 + l16;
        const int akb  = k0 + lhi * 8;
#pragma unroll
        for (int r = 0; r < 8; ++r) {
            const int kk = akb + 2 * (r & 3) + (r >> 2) * 16;
            const float2 f = *(const float2*)&A[arow * DD + kk];
            Af.u[r] = pack2bf(f.x, f.y);
        }
        // B fragment: 32x16 bf16, N = l16, K layout per ISA table
        const int bcol = n0 + l16;
        const int bkb  = k0 + lhi * 8;
#pragma unroll
        for (int r = 0; r < 8; ++r) {
            const int kk = bkb + 2 * (r & 3) + (r >> 2) * 16;
            Bf.u[r] = pack2bf(W[kk * DD + bcol], W[(kk + 1) * DD + bcol]);
        }
        acc = __builtin_amdgcn_wmma_f32_16x16x32_bf16(
                  false, Af.v, false, Bf.v, (short)0, acc, false, false);
    }
    const int col = n0 + l16;
    const float b = bias[col];
#pragma unroll
    for (int p = 0; p < 8; ++p) {
        const int row = m0 + p + lhi * 8;
        out[row * DD + col] = (acc[p] + b) * scale;
    }
}

// ---------------------------------------------------------------------------
// Kernel 2: qk[bi, in, h] = sum_d Qp[bi, 64h+d] * Wk[in, 64h+d]   (bf16 out)
// Per-head GEMM  M=512(bi), N=768(in), K=64.  grid = (32, 48, 12), block = 32.
// Output layout (per bi): pair-interleaved B matrix, u32 index = (in/2)*16 + h,
// with (in even -> [15:0], in odd -> [31:16]) so each WMMA B register is one
// aligned 32-bit LDS load in the attention kernel.
// ---------------------------------------------------------------------------
__global__ __launch_bounds__(32) void qk_head(
    const float* __restrict__ Qp, const float* __restrict__ Wk,
    unsigned short* __restrict__ qkout)
{
    const int m0 = blockIdx.x * 16;          // bi tile
    const int n0 = blockIdx.y * 16;          // 'in' tile
    const int h  = blockIdx.z;
    const int lane = threadIdx.x & 31;
    const int lhi = (lane >> 4) & 1;
    const int l16 = lane & 15;

    v8f acc = {};
#pragma unroll
    for (int k0 = 0; k0 < 64; k0 += 32) {
        BF16Frag Af, Bf;
        const int arow = m0 + l16;
        const int akb  = k0 + lhi * 8;
#pragma unroll
        for (int r = 0; r < 8; ++r) {
            const int kk = akb + 2 * (r & 3) + (r >> 2) * 16;
            const float2 f = *(const float2*)&Qp[arow * DD + h * 64 + kk];
            Af.u[r] = pack2bf(f.x, f.y);
        }
        // B[k=d][n=in] = Wk[in, 64h+d]: consecutive k is row-contiguous
        const int nin = n0 + l16;
        const int bkb = k0 + lhi * 8;
#pragma unroll
        for (int r = 0; r < 8; ++r) {
            const int kk = bkb + 2 * (r & 3) + (r >> 2) * 16;
            const float2 f = *(const float2*)&Wk[nin * DD + h * 64 + kk];
            Bf.u[r] = pack2bf(f.x, f.y);
        }
        acc = __builtin_amdgcn_wmma_f32_16x16x32_bf16(
                  false, Af.v, false, Bf.v, (short)0, acc, false, false);
    }
    const int nin = n0 + l16;
#pragma unroll
    for (int p = 0; p < 8; ++p) {
        const int row = m0 + p + lhi * 8;   // bi
        // pair-interleaved layout: halves index = ((in/2)*16 + h)*2 + (in&1)
        qkout[(size_t)row * (DD * 16) + (((nin >> 1) * 16 + h) << 1) + (nin & 1)]
            = f2bf(acc[p]);
    }
}

// ---------------------------------------------------------------------------
// Kernel 3: fused attention core.  One block per (b,i) row.  block = 256 (8 waves).
//   scores[j,h] = (key[j,:] . qk[:,h] + qb[h]) * mask[j]        (WMMA, waves 0-3)
//   (waves 4-7 prefetch the value tile into cache meanwhile)
//   softmax over j                                              (16 threads)
//   vbar[h,:]   = attn[h,:] @ value[:, :]                       (WMMA, 8 waves)
//   ctx[64h+d]  = vbar[h,:] . Wv[:, 64h+d] + bv                 (VALU)
// ---------------------------------------------------------------------------
__global__ __launch_bounds__(256) void attn_core(
    const float* __restrict__ key,      // [BI, LL, DD]
    const float* __restrict__ value,    // [BI, LL, DD]
    const float* __restrict__ mask,     // [BI, LL]
    const float* __restrict__ Qp,       // [BI, DD] (already scaled by 1/8)
    const unsigned short* __restrict__ qk, // [BI][DD/2][16][2] bf16 (pair-interleaved)
    const float* __restrict__ bk,       // [DD]
    const float* __restrict__ Wv,       // [DD, DD]
    const float* __restrict__ bv,       // [DD]
    float* __restrict__ ctx)            // [BI, DD]
{
    __shared__ __align__(16) unsigned short sKey[LL * DD];   // 96 KB bf16 key tile
    __shared__ __align__(16) unsigned short sQk[DD * 16];    // 24 KB bf16 qk (B matrix)
    __shared__ __align__(16) float          sS[LL * 16];     //  4 KB scores [j][h]
    __shared__ __align__(16) unsigned short sAttn[16 * LL];  //  2 KB attn (A matrix) [h][j]
    __shared__ __align__(16) float          sVbar[16 * DD];  // 48 KB vbar f32
    __shared__ float sQb[16];
    __shared__ float sMask[LL];

    const int bi   = blockIdx.x;
    const int t    = threadIdx.x;
    const int lane = t & 31;
    const int wave = t >> 5;
    const int lhi  = (lane >> 4) & 1;
    const int l16  = lane & 15;

    const float* keyR = key   + (size_t)bi * LL * DD;
    const float* valR = value + (size_t)bi * LL * DD;

    // ---- stage key tile -> bf16 LDS (coalesced 16B global loads) ----
    const float4* k4 = (const float4*)keyR;
    for (int idx = t; idx < LL * DD / 4; idx += 256) {
        const float4 v = k4[idx];
        uint2 p;
        p.x = pack2bf(v.x, v.y);
        p.y = pack2bf(v.z, v.w);
        *(uint2*)&sKey[idx * 4] = p;
    }
    // ---- stage qk (already bf16, pair-interleaved); zero padded heads 12-15 ----
    {
        const unsigned int* g = (const unsigned int*)(qk + (size_t)bi * DD * 16);
        unsigned int* s32 = (unsigned int*)sQk;
        for (int idx = t; idx < DD * 16 / 2; idx += 256)
            s32[idx] = ((idx & 15) < HH) ? g[idx] : 0u;
    }
    if (t < LL) sMask[t] = mask[(size_t)bi * LL + t];
    // ---- qb[h] = Qp[bi, h-slice] . bk[h-slice] ----
    if (t < 16) {
        float a = 0.0f;
        if (t < HH) {
            const float* qp = &Qp[(size_t)bi * DD + t * 64];
            const float* bb = &bk[t * 64];
#pragma unroll 8
            for (int d = 0; d < 64; ++d) a += qp[d] * bb[d];
        }
        sQb[t] = a;
    }
    __syncthreads();

    if (wave < 4) {
        // ---- scores: [64 x 768] @ [768 x 16] via WMMA, waves 0-3 own j-tiles ----
        const int m0 = wave * 16;
        const unsigned int* s32 = (const unsigned int*)sQk;
        v8f acc = {};
        for (int k0 = 0; k0 < DD; k0 += 32) {
            BF16Frag Af, Bf;
            const int arow = m0 + l16;
            const int akb  = k0 + lhi * 8;
#pragma unroll
            for (int r = 0; r < 8; ++r) {
                const int kk = akb + 2 * (r & 3) + (r >> 2) * 16;
                Af.u[r] = *(const unsigned int*)&sKey[arow * DD + kk];
            }
            // pair-interleaved B: one aligned u32 per register
            const int kb2 = (k0 >> 1) + lhi * 4;
#pragma unroll
            for (int r = 0; r < 8; ++r)
                Bf.u[r] = s32[(kb2 + (r & 3) + (r >> 2) * 8) * 16 + l16];
            acc = __builtin_amdgcn_wmma_f32_16x16x32_bf16(
                      false, Af.v, false, Bf.v, (short)0, acc, false, false);
        }
#pragma unroll
        for (int p = 0; p < 8; ++p) {
            const int j = m0 + p + lhi * 8;
            sS[j * 16 + l16] = (acc[p] + sQb[l16]) * sMask[j];
        }
    } else {
        // ---- waves 4-7: prefetch value tile (192 KB) into cache ----
        const char* vp = (const char*)valR;
        const int local = t - 128;                    // 0..127
        for (int i = local; i < (LL * DD * 4) / 128; i += 128)
            __builtin_prefetch(vp + i * 128, 0, 3);
    }
    __syncthreads();

    // ---- softmax over j (per head), write attn as bf16 A-matrix [h][j] ----
    if (t < 16) {
        if (t < HH) {
            float m = -1e30f;
            for (int j = 0; j < LL; ++j) m = fmaxf(m, sS[j * 16 + t]);
            float sum = 0.0f;
            for (int j = 0; j < LL; ++j) sum += __expf(sS[j * 16 + t] - m);
            const float inv = 1.0f / sum;
            for (int j = 0; j < LL; ++j)
                sAttn[t * LL + j] = f2bf(__expf(sS[j * 16 + t] - m) * inv);
        } else {
            for (int j = 0; j < LL; ++j) sAttn[t * LL + j] = 0;
        }
    }
    __syncthreads();

    // ---- vbar: [16 x 64] @ [64 x 768] via WMMA; value streamed from HBM once ----
    for (int nt = wave; nt < DD / 16; nt += 8) {
        const int n0 = nt * 16;
        v8f acc = {};
#pragma unroll
        for (int k0 = 0; k0 < LL; k0 += 32) {
            BF16Frag Af, Bf;
            const int akb = k0 + lhi * 8;
#pragma unroll
            for (int r = 0; r < 8; ++r) {
                const int kk = akb + 2 * (r & 3) + (r >> 2) * 16;
                Af.u[r] = *(const unsigned int*)&sAttn[l16 * LL + kk];
            }
            const int col = n0 + l16;
            const int bkb = k0 + lhi * 8;
#pragma unroll
            for (int r = 0; r < 8; ++r) {
                const int j = bkb + 2 * (r & 3) + (r >> 2) * 16;
                Bf.u[r] = pack2bf(valR[j * DD + col], valR[(j + 1) * DD + col]);
            }
            acc = __builtin_amdgcn_wmma_f32_16x16x32_bf16(
                      false, Af.v, false, Bf.v, (short)0, acc, false, false);
        }
        const int col = n0 + l16;
#pragma unroll
        for (int p = 0; p < 8; ++p) {
            const int h = p + lhi * 8;
            sVbar[h * DD + col] = acc[p];
        }
    }
    __syncthreads();

    // ---- ctx[o] = vbar[o/64, :] . Wv[:, o] + bv[o]  (block-diagonal proj) ----
    float* ctxR = ctx + (size_t)bi * DD;
#pragma unroll
    for (int s = 0; s < 3; ++s) {
        const int o = t + 256 * s;          // 0..767, lanes contiguous in o
        const float* vb = &sVbar[(o >> 6) * DD];
        float a = bv[o];
#pragma unroll 8
        for (int in = 0; in < DD; ++in) a += vb[in] * Wv[in * DD + o];
        ctxR[o] = a;
    }
}

// ---------------------------------------------------------------------------
extern "C" void kernel_launch(void* const* d_in, const int* in_sizes, int n_in,
                              void* d_out, int out_size, void* d_ws, size_t ws_size,
                              hipStream_t stream) {
    (void)in_sizes; (void)n_in; (void)out_size; (void)ws_size;
    const float* key   = (const float*)d_in[0];
    const float* value = (const float*)d_in[1];
    const float* query = (const float*)d_in[2];
    const float* mask  = (const float*)d_in[3];
    const float* Wk    = (const float*)d_in[4];
    const float* bk    = (const float*)d_in[5];
    const float* Wv    = (const float*)d_in[6];
    const float* bv    = (const float*)d_in[7];
    const float* Wq    = (const float*)d_in[8];
    const float* bq    = (const float*)d_in[9];
    const float* Wo    = (const float*)d_in[10];
    const float* bo    = (const float*)d_in[11];
    float* out = (float*)d_out;

    // workspace layout: Qp fp32 @0 (1.5MB), qk bf16 @2MB (12MB), ctx fp32 @16MB (1.5MB)
    float*          Qp  = (float*)d_ws;
    unsigned short* qkw = (unsigned short*)((char*)d_ws + (size_t)(2u << 20));
    float*          ctx = (float*)((char*)d_ws + (size_t)(16u << 20));

    dim3 gGemm(BI / 16, DD / 16);        // 32 x 48
    // 1) Qp = (query @ Wq + bq) * (1/sqrt(64))
    gemm512_bias<<<gGemm, 32, 0, stream>>>(query, Wq, bq, Qp, 0.125f);
    // 2) qk[bi, in, h]  (pair-interleaved bf16)
    dim3 gQk(BI / 16, DD / 16, HH);      // 32 x 48 x 12
    qk_head<<<gQk, 32, 0, stream>>>(Qp, Wk, qkw);
    // 3) fused attention core -> ctx
    attn_core<<<BI, 256, 0, stream>>>(key, value, mask, Qp, qkw, bk, Wv, bv, ctx);
    // 4) out = ctx @ Wo + bo
    gemm512_bias<<<gGemm, 32, 0, stream>>>(ctx, Wo, bo, out, 1.0f);
}
